// GATlayer_IMP8_Hp_20529943674899
// MI455X (gfx1250) — matrix operand order repaired and checked by reading, not compiled, not run on hardware
//
#include <hip/hip_runtime.h>
#include <hip/hip_bf16.h>

// GAT layer for gfx1250 (MI455X): bf16 WMMA GEMMs, async-LDS tile staging,
// flash-style fused masked softmax aggregation, f32 epilogue.
//
// Sizes fixed by the reference:
#define NN    4096      // nodes
#define FIN   128       // input features
#define NH    8         // heads
#define FO    64        // features per head
#define HF    512       // NH*FO

typedef __attribute__((ext_vector_type(16))) __bf16 v16bf;
typedef __attribute__((ext_vector_type(8)))  float  v8f;

union Frag { unsigned u[8]; v16bf v; };

__device__ __forceinline__ unsigned pack_bf16(float a, float b) {
    union { __bf16 h[2]; unsigned u; } p;
    p.h[0] = (__bf16)a; p.h[1] = (__bf16)b;
    return p.u;
}

// A-fragment packed-pair index for 16-bit 16x32 A matrix (contiguous K, matches B):
// lanes 0-15: VGPR0..7 -> K=0..15
// lanes16-31: VGPR0..7 -> K=16..31
__device__ __forceinline__ int a_pair(int v, int half) {
    return v + (half << 3);   // pair index, k = 2*pair
}
// B-fragment (32x16, 16-bit): lanes 0-15 col N, K=2v,2v+1; lanes 16-31 K=16+2v
__device__ __forceinline__ int b_pair(int v, int half) {
    return v + (half << 3);
}

// Async memory->LDS copy, 16 bytes per lane (VGLOBAL async encoding, GV mode).
// Tracked by ASYNCcnt; must s_wait_asynccnt before LDS reads.
__device__ __forceinline__ void async_copy16(unsigned lds_byte_off, const void* gptr) {
    unsigned long long ga = (unsigned long long)gptr;
    asm volatile("global_load_async_to_lds_b128 %0, %1, off"
                 :: "v"(lds_byte_off), "v"(ga) : "memory");
}
__device__ __forceinline__ void wait_async0() {
    asm volatile("s_wait_asynccnt 0x0" ::: "memory");
}

// ---------------------------------------------------------------------------
// Kernel 0: f32 -> packed bf16 pairs for x, W_proj, W_skip (DMA-able operands).
// ---------------------------------------------------------------------------
#define XPAIRS  (NN * FIN / 2)        // 262144
#define WPAIRS  (HF * FIN / 2)        // 32768
__global__ __launch_bounds__(256)
void cvt_bf16_kernel(const float* __restrict__ x,
                     const float* __restrict__ wp,
                     const float* __restrict__ wsk,
                     unsigned* __restrict__ xb,
                     unsigned* __restrict__ wpb,
                     unsigned* __restrict__ wsb)
{
    const int t = blockIdx.x * 256 + threadIdx.x;   // pair index
    const float* src; unsigned* dst; int off;
    if (t < XPAIRS)               { src = x;   dst = xb;  off = t; }
    else if (t < XPAIRS + WPAIRS) { src = wp;  dst = wpb; off = t - XPAIRS; }
    else                          { src = wsk; dst = wsb; off = t - XPAIRS - WPAIRS; }
    const float2 v = *(const float2*)(src + (size_t)off * 2);
    dst[off] = pack_bf16(v.x, v.y);
}

// ---------------------------------------------------------------------------
// Kernel 1: proj = x @ Wp^T (stored transposed, bf16) and skip = x @ Ws^T (f32)
// M=4096 rows, 1024 output cols (0-511 proj, 512-1023 skip), K=128.
// Block = 128 thr (4 waves), 64x64 output tile, K-steps of 32.
// Tiles staged into LDS with GLOBAL_LOAD_ASYNC_TO_LDS_B128 (ASYNCcnt path).
// ---------------------------------------------------------------------------
__global__ __launch_bounds__(128)
void gat_proj_kernel(const unsigned* __restrict__ xb,   // [NN][FIN/2] bf16 pairs
                     const unsigned* __restrict__ wpb,  // [HF][FIN/2]
                     const unsigned* __restrict__ wsb,  // [HF][FIN/2]
                     unsigned*       __restrict__ projT_u,  // [HF][NN/2] pairs
                     float*          __restrict__ skip)     // [NN][HF]
{
    __shared__ unsigned xlds[64 * 16];   // 64 rows x 16 bf16-pairs (K=32)
    __shared__ unsigned wlds[64 * 16];

    const int nb   = blockIdx.x;            // row tile: 64 nodes
    const int ob   = blockIdx.y;            // col tile: 0-7 proj, 8-15 skip
    const int tid  = threadIdx.x;
    const int lane = tid & 31, wave = tid >> 5;
    const int half = lane >> 4, l16 = lane & 15;

    const bool is_proj    = (ob < 8);
    const unsigned* Wb    = is_proj ? wpb : wsb;
    const int   ocol      = (ob & 7) * 64;   // col offset within 512
    const int   row0      = nb * 64;

    v8f acc[4] = {};

    for (int kb = 0; kb < 4; ++kb) {
        const int k0p = kb * 16;             // pair offset within a row
        // Async DMA the 64x32 bf16 tiles: 256 x 16B chunks per tile,
        // 2 chunks per thread per tile.
        #pragma unroll
        for (int p = 0; p < 2; ++p) {
            const int c = tid + p * 128;     // 0..255
            const int r = c >> 2, q = (c & 3) * 4;
            async_copy16((unsigned)(size_t)&xlds[r * 16 + q],
                         xb + (size_t)(row0 + r) * (FIN / 2) + k0p + q);
            async_copy16((unsigned)(size_t)&wlds[r * 16 + q],
                         Wb + (size_t)(ocol + r) * (FIN / 2) + k0p + q);
        }
        wait_async0();
        __syncthreads();

        // A fragment: this wave's 16 node rows.
        Frag a;
        const int m = wave * 16 + l16;
        #pragma unroll
        for (int v = 0; v < 8; ++v) a.u[v] = xlds[m * 16 + a_pair(v, half)];

        #pragma unroll
        for (int g = 0; g < 4; ++g) {
            Frag b;
            const int oc = g * 16 + l16;
            #pragma unroll
            for (int v = 0; v < 8; ++v) b.u[v] = wlds[oc * 16 + b_pair(v, half)];
            acc[g] = __builtin_amdgcn_wmma_f32_16x16x32_bf16(
                         false, a.v, false, b.v, (short)0, acc[g], false, false);
        }
        __syncthreads();
    }

    // Epilogue. C/D layout: lane<16 -> N=lane, VGPR r -> M=r; lane>=16 -> M=r+8.
    const int mrow = row0 + wave * 16 + half * 8;   // 8 consecutive node rows
    if (is_proj) {
        #pragma unroll
        for (int g = 0; g < 4; ++g) {
            const int o = ocol + g * 16 + l16;
            uint4 pk;
            pk.x = pack_bf16(acc[g][0], acc[g][1]);
            pk.y = pack_bf16(acc[g][2], acc[g][3]);
            pk.z = pack_bf16(acc[g][4], acc[g][5]);
            pk.w = pack_bf16(acc[g][6], acc[g][7]);
            *(uint4*)(projT_u + (size_t)o * (NN / 2) + (mrow >> 1)) = pk;
        }
    } else {
        #pragma unroll
        for (int g = 0; g < 4; ++g) {
            const int o = ocol + g * 16 + l16;
            #pragma unroll
            for (int r = 0; r < 8; ++r)
                skip[(size_t)(mrow + r) * HF + o] = acc[g][r];
        }
    }
}

// ---------------------------------------------------------------------------
// Kernel 2: s_src[h][n] = sum_d proj[n,h,d]*a_src[h,d]  (same for s_tgt).
// ---------------------------------------------------------------------------
__global__ __launch_bounds__(256)
void gat_scores_kernel(const __bf16* __restrict__ projT,   // [HF][NN]
                       const float*  __restrict__ a_src,   // [NH*FO]
                       const float*  __restrict__ a_tgt,
                       float* __restrict__ s_src,          // [NH][NN]
                       float* __restrict__ s_tgt)
{
    const int t = blockIdx.x * 256 + threadIdx.x;   // 0..NN*NH-1
    const int n = t >> 3, h = t & 7;
    float as = 0.f, at = 0.f;
    #pragma unroll 8
    for (int d = 0; d < FO; ++d) {
        const float p = (float)projT[(size_t)(h * FO + d) * NN + n];
        as += p * a_src[h * FO + d];
        at += p * a_tgt[h * FO + d];
    }
    s_src[h * NN + n] = as;
    s_tgt[h * NN + n] = at;
}

// ---------------------------------------------------------------------------
// Kernel 3: per (head,row) softmax stats: rmax + 1/sum(exp). adj L2-resident.
// One wave per row; grid (NN/8, NH), block 256 (8 waves).
// ---------------------------------------------------------------------------
__global__ __launch_bounds__(256)
void gat_rowstats_kernel(const float* __restrict__ adj,
                         const float* __restrict__ s_src,
                         const float* __restrict__ s_tgt,
                         float* __restrict__ rmax,
                         float* __restrict__ rinv)
{
    const int lane = threadIdx.x & 31, wave = threadIdx.x >> 5;
    const int i = blockIdx.x * 8 + wave;
    const int h = blockIdx.y;
    const float ss = s_src[h * NN + i];
    const float* st   = s_tgt + h * NN;
    const float* arow = adj + (size_t)i * NN;

    float m = -3.4e38f;
    for (int j = lane; j < NN; j += 32) {
        float e = ss + st[j];
        e = e > 0.f ? e : 0.2f * e;            // leaky_relu(0.2)
        e += arow[j];                          // additive mask
        m = fmaxf(m, e);
    }
    #pragma unroll
    for (int off = 16; off > 0; off >>= 1) m = fmaxf(m, __shfl_xor(m, off, 32));

    float s = 0.f;
    for (int j = lane; j < NN; j += 32) {
        float e = ss + st[j];
        e = e > 0.f ? e : 0.2f * e;
        e += arow[j];
        s += __expf(e - m);
    }
    #pragma unroll
    for (int off = 16; off > 0; off >>= 1) s += __shfl_xor(s, off, 32);

    if (lane == 0) {
        rmax[h * NN + i] = m;
        rinv[h * NN + i] = 1.f / s;
    }
}

// ---------------------------------------------------------------------------
// Kernel 4: O = softmax(score) @ V, fused with skip + bias + ELU.
// Grid (NN/64, NH), block 128 (4 waves). Each wave: 16 rows x 64 d.
// P fragment built in registers (scores are rank-1 + adj), V from projT.
// ---------------------------------------------------------------------------
__global__ __launch_bounds__(128)
void gat_attn_kernel(const float*    __restrict__ adj,
                     const unsigned* __restrict__ projT_u,  // [HF][NN/2] pairs
                     const float*    __restrict__ s_src,
                     const float*    __restrict__ s_tgt,
                     const float*    __restrict__ rmax,
                     const float*    __restrict__ rinv,
                     const float*    __restrict__ skip,
                     const float*    __restrict__ bias,
                     float*          __restrict__ out)
{
    const int tid  = threadIdx.x;
    const int lane = tid & 31, wave = tid >> 5;
    const int half = lane >> 4, l16 = lane & 15;
    const int h    = blockIdx.y;
    const int i0   = blockIdx.x * 64 + wave * 16;   // wave row base
    const int row  = i0 + l16;                      // this lane's A-frag row

    const float ss  = s_src[h * NN + row];
    const float mrw = rmax[h * NN + row];
    const float rin = rinv[h * NN + row];
    const float* st   = s_tgt + h * NN;
    const float* arow = adj + (size_t)row * NN;

    v8f acc[4] = {};

    for (int j0 = 0; j0 < NN; j0 += 32) {
        // Keep the L2-resident adj stream ahead of the VALU-bound exp work.
        if (j0 + 256 < NN) __builtin_prefetch(arow + j0 + 256, 0, 1);

        // Build normalized-probability A fragment (16 rows x 32 cols, bf16).
        Frag a;
        #pragma unroll
        for (int v = 0; v < 8; ++v) {
            const int j = j0 + a_pair(v, half) * 2;
            const float2 stv = *(const float2*)(st + j);
            const float2 av  = *(const float2*)(arow + j);
            float e0 = ss + stv.x; e0 = e0 > 0.f ? e0 : 0.2f * e0;
            float e1 = ss + stv.y; e1 = e1 > 0.f ? e1 : 0.2f * e1;
            const float p0 = __expf(e0 + av.x - mrw) * rin;
            const float p1 = __expf(e1 + av.y - mrw) * rin;
            a.u[v] = pack_bf16(p0, p1);
        }
        // V fragments from transposed projection (contiguous K-pairs).
        const int jp = (j0 >> 1) + (half << 3);
        #pragma unroll
        for (int g = 0; g < 4; ++g) {
            Frag b;
            const unsigned* vrow =
                projT_u + (size_t)(h * FO + g * 16 + l16) * (NN / 2) + jp;
            #pragma unroll
            for (int v = 0; v < 8; ++v) b.u[v] = vrow[v];
            acc[g] = __builtin_amdgcn_wmma_f32_16x16x32_bf16(
                         false, a.v, false, b.v, (short)0, acc[g], false, false);
        }
    }

    // Epilogue: + skip + bias, ELU, store f32.
    const int nbase = i0 + half * 8;
    #pragma unroll
    for (int g = 0; g < 4; ++g) {
        const int o = h * FO + g * 16 + l16;
        const float bi = bias[o];
        #pragma unroll
        for (int r = 0; r < 8; ++r) {
            const int n = nbase + r;
            float v = acc[g][r] + skip[(size_t)n * HF + o] + bi;
            v = v > 0.f ? v : (__expf(v) - 1.f);   // ELU
            out[(size_t)n * HF + o] = v;
        }
    }
}

// ---------------------------------------------------------------------------
extern "C" void kernel_launch(void* const* d_in, const int* in_sizes, int n_in,
                              void* d_out, int out_size, void* d_ws, size_t ws_size,
                              hipStream_t stream) {
    const float* x     = (const float*)d_in[0];   // [1,4096,128]
    const float* adj   = (const float*)d_in[1];   // [4096,4096]
    const float* Wp    = (const float*)d_in[2];   // [512,128]
    const float* a_src = (const float*)d_in[3];   // [1,8,64]
    const float* a_tgt = (const float*)d_in[4];   // [1,8,64]
    const float* Wsk   = (const float*)d_in[5];   // [512,128]
    const float* bias  = (const float*)d_in[6];   // [512]
    float* out = (float*)d_out;

    char* ws = (char*)d_ws;
    unsigned* projT_u = (unsigned*)ws;                       // 4 MB  bf16 [512][4096]
    float*    skip    = (float*)(ws + (4u  << 20));          // 8 MB  f32  [4096][512]
    float*    s_src   = (float*)(ws + (12u << 20));          // 4x128 KB stats
    float*    s_tgt   = s_src + NN * NH;
    float*    rmax    = s_tgt + NN * NH;
    float*    rinv    = rmax + NN * NH;
    unsigned* xb      = (unsigned*)(ws + (13u << 20));       // 1 MB   bf16 x
    unsigned* wpb     = (unsigned*)(ws + (14u << 20));       // 128 KB bf16 W_proj
    unsigned* wsb     = wpb + WPAIRS;                        // 128 KB bf16 W_skip

    cvt_bf16_kernel<<<dim3((XPAIRS + 2 * WPAIRS) / 256), dim3(256), 0, stream>>>(
        x, Wp, Wsk, xb, wpb, wsb);

    gat_proj_kernel<<<dim3(NN / 64, 16), dim3(128), 0, stream>>>(
        xb, wpb, wsb, projT_u, skip);

    gat_scores_kernel<<<dim3(NN * NH / 256), dim3(256), 0, stream>>>(
        (const __bf16*)projT_u, a_src, a_tgt, s_src, s_tgt);

    gat_rowstats_kernel<<<dim3(NN / 8, NH), dim3(256), 0, stream>>>(
        adj, s_src, s_tgt, rmax, rinv);

    gat_attn_kernel<<<dim3(NN / 64, NH), dim3(128), 0, stream>>>(
        adj, projT_u, s_src, s_tgt, rmax, rinv, skip, bias, out);

    // Reference returns (out, adj): copy adj into the second tuple slot.
    hipMemcpyAsync(out + (size_t)NN * HF, adj, (size_t)NN * NN * sizeof(float),
                   hipMemcpyDeviceToDevice, stream);
}
